// SwinTransformerBlock_37142877176249
// MI455X (gfx1250) — compile-verified, matrix operand order checked by reading, not run on hardware
//
#include <hip/hip_runtime.h>
#include <hip/hip_bf16.h>

typedef __attribute__((ext_vector_type(16))) __bf16 v16bf;
typedef __attribute__((ext_vector_type(8)))  float  v8f;
typedef __attribute__((ext_vector_type(4)))  int    v4i;

#define BATCH 32
#define HH 56
#define CC 192
#define WS 7
#define SSH 3
#define NHEAD 6
#define HD 32
#define NTOK 49          // WS*WS
#define NWIN 64          // (H/WS)^2
#define NWINS 2048       // B*NW
#define MROWS 100352     // NWINS*NTOK
#define QKV_TSZ 19267584 // NWINS*NHEAD*NTOK*HD elements (per q/k/v tensor)
#define SCALE_Q 0.17677669529663687f

#define GASP __attribute__((address_space(1)))
#define LASP __attribute__((address_space(3)))

// ---------------------------------------------------------------------------
// CDNA5 async global->LDS copy (16B per lane), ASYNCcnt-tracked.
// ---------------------------------------------------------------------------
__device__ inline void async_copy16(const __bf16* g, __bf16* l) {
#if __has_builtin(__builtin_amdgcn_global_load_async_to_lds_b128)
    __builtin_amdgcn_global_load_async_to_lds_b128(
        (GASP v4i*)(GASP void*)(void*)const_cast<__bf16*>(g),
        (LASP v4i*)(LASP void*)(void*)l, 0, 0);
#else
    *(uint4*)l = *(const uint4*)g;
#endif
}

#if __has_builtin(__builtin_amdgcn_s_wait_asynccnt)
#define WAIT_ASYNC(n) __builtin_amdgcn_s_wait_asynccnt(n)
#else
#define WAIT_ASYNC(n) asm volatile("s_wait_asynccnt %0" ::"i"(n) : "memory")
#endif

// ---------------------------------------------------------------------------
// Fragment loader: ISA 16-bit A-matrix layout (lane<16: row, K = {0..7,16..23};
// lane>=16: same row, K = {8..15,24..31}).  Caller pre-adds row*ld + hi*8 + k0.
// ---------------------------------------------------------------------------
__device__ inline v16bf load_frag(const __bf16* p) {
    v16bf f;
#pragma unroll
    for (int i = 0; i < 8; ++i) { f[i] = p[i]; f[i + 8] = p[i + 16]; }
    return f;
}

__device__ inline v8f wmma_bf16(v16bf a, v16bf b, v8f c) {
    return __builtin_amdgcn_wmma_f32_16x16x32_bf16(false, a, false, b,
                                                   (short)0, c, false, false);
}

// ---------------------------------------------------------------------------
// Weight convert + transpose: w (K x N, f32) -> wt (N x K, bf16)
// ---------------------------------------------------------------------------
__global__ void conv_wt_kernel(const float* __restrict__ w, __bf16* __restrict__ wt,
                               int K, int N) {
    int idx = blockIdx.x * 256 + threadIdx.x;
    if (idx < K * N) {
        int k = idx / N, n = idx % N;
        wt[(size_t)n * K + k] = (__bf16)w[idx];
    }
}

// ---------------------------------------------------------------------------
// LayerNorm (wave per row).  shifted=1: fused roll(-SS,-SS)+window partition.
// ---------------------------------------------------------------------------
__global__ __launch_bounds__(256) void ln_kernel(const float* __restrict__ x,
                                                 const float* __restrict__ g,
                                                 const float* __restrict__ bta,
                                                 __bf16* __restrict__ dst,
                                                 int shifted) {
    int row  = blockIdx.x * 8 + (threadIdx.x >> 5);
    int lane = threadIdx.x & 31;
    const float* src;
    if (shifted) {
        int win = row / NTOK, n = row % NTOK;
        int b = win >> 6, wrem = win & 63, wi = wrem >> 3, wj = wrem & 7;
        int i = n / WS, j = n % WS;
        int hs = (wi * WS + i + SSH) % HH;
        int ws = (wj * WS + j + SSH) % HH;
        src = x + ((size_t)b * (HH * HH) + hs * HH + ws) * CC;
    } else {
        src = x + (size_t)row * CC;
    }
    float vals[6];
    float s = 0.f;
#pragma unroll
    for (int t = 0; t < 6; ++t) { vals[t] = src[lane + t * 32]; s += vals[t]; }
#pragma unroll
    for (int o = 16; o; o >>= 1) s += __shfl_xor(s, o, 32);
    float mu = s * (1.0f / CC);
    float vv = 0.f;
#pragma unroll
    for (int t = 0; t < 6; ++t) { float d = vals[t] - mu; vv += d * d; }
#pragma unroll
    for (int o = 16; o; o >>= 1) vv += __shfl_xor(vv, o, 32);
    float rstd = rsqrtf(vv * (1.0f / CC) + 1e-5f);
    __bf16* d = dst + (size_t)row * CC;
#pragma unroll
    for (int t = 0; t < 6; ++t) {
        int c = lane + t * 32;
        d[c] = (__bf16)((vals[t] - mu) * rstd * g[c] + bta[c]);
    }
}

// ---------------------------------------------------------------------------
// bf16 WMMA GEMM: C = A(MxK) * Bt(NxK)^T + bias, 4 fused epilogues.
// Block = 256 thr (8 waves); block tile 128x64; K step 32.
// A/B tiles double-buffered in LDS via async global->LDS copies (ASYNCcnt).
// ---------------------------------------------------------------------------
enum { EPI_QKV = 0, EPI_PROJ = 1, EPI_GELU = 2, EPI_OUT = 3 };

template <int EPI>
__global__ __launch_bounds__(256) void gemm_bf16_kernel(
        const __bf16* __restrict__ A, const __bf16* __restrict__ Bt,
        const float* __restrict__ bias, int M, int N, int K,
        __bf16* __restrict__ outb,          // QKV: q base (k,v follow); GELU: h1
        float* __restrict__ outf,           // PROJ: x1 ; OUT: d_out
        const float* __restrict__ res) {    // PROJ: shortcut x ; OUT: x1
    __shared__ __bf16 As[2][128 * 32];
    __shared__ __bf16 Bs[2][64 * 32];

    int tid  = threadIdx.x;
    int wave = tid >> 5, lane = tid & 31;
    int l15 = lane & 15, hi = lane >> 4;
    int mblk = blockIdx.x * 128;
    int n0   = blockIdx.y * 64;

    // Stage one K-slab (A: 128x32, B: 64x32) into LDS buffer `buf`.
    // 3 async instructions per wave per stage.
    auto stage = [&](int buf, int k0) {
#pragma unroll
        for (int c = 0; c < 2; ++c) {
            int idx = tid + c * 256;          // 512 chunks of 16B for A
            int row = idx >> 2, seg = idx & 3;
            async_copy16(A + (size_t)(mblk + row) * K + k0 + seg * 8,
                         &As[buf][row * 32 + seg * 8]);
        }
        {
            int row = tid >> 2, seg = tid & 3; // 256 chunks of 16B for B
            async_copy16(Bt + (size_t)(n0 + row) * K + k0 + seg * 8,
                         &Bs[buf][row * 32 + seg * 8]);
        }
    };

    int nk = K >> 5;
    stage(0, 0);
    v8f acc[4] = {};
    for (int kk = 0; kk < nk; ++kk) {
        if (kk + 1 < nk) {
            stage((kk + 1) & 1, (kk + 1) << 5);
            WAIT_ASYNC(3);   // retire stage kk; stage kk+1 stays in flight
        } else {
            WAIT_ASYNC(0);
        }
        __syncthreads();
        int buf = kk & 1;
        v16bf af = load_frag(&As[buf][(wave * 16 + l15) * 32 + hi * 8]);
#pragma unroll
        for (int t = 0; t < 4; ++t) {
            v16bf bfr = load_frag(&Bs[buf][(t * 16 + l15) * 32 + hi * 8]);
            acc[t] = wmma_bf16(af, bfr, acc[t]);
        }
        __syncthreads();     // buffer `buf` may be recycled next iteration
    }

#pragma unroll
    for (int t = 0; t < 4; ++t) {
        int col = n0 + t * 16 + l15;
        float bv = bias[col];
#pragma unroll
        for (int r = 0; r < 8; ++r) {
            int row = mblk + wave * 16 + hi * 8 + r;
            float v = acc[t][r] + bv;
            if constexpr (EPI == EPI_QKV) {
                int which = col / CC, cw = col % CC;
                int head = cw >> 5, d = cw & 31;
                if (which == 0) v *= SCALE_Q;
                int win = row / NTOK, n = row % NTOK;
                size_t dst = ((size_t)which * (NWINS * NHEAD) +
                              (size_t)win * NHEAD + head) * (NTOK * HD) + n * HD + d;
                outb[dst] = (__bf16)v;
            } else if constexpr (EPI == EPI_PROJ) {
                int win = row / NTOK, n = row % NTOK;
                int b = win >> 6, wrem = win & 63, wi = wrem >> 3, wj = wrem & 7;
                int i = n / WS, j = n % WS;
                int h = (wi * WS + i + SSH) % HH;
                int w = (wj * WS + j + SSH) % HH;
                size_t idx = ((size_t)b * (HH * HH) + h * HH + w) * CC + col;
                outf[idx] = res[idx] + v;
            } else if constexpr (EPI == EPI_GELU) {
                float gl = 0.5f * v * (1.0f + erff(v * 0.7071067811865476f));
                outb[(size_t)row * N + col] = (__bf16)gl;
            } else { // EPI_OUT
                size_t idx = (size_t)row * N + col;
                outf[idx] = res[idx] + v;
            }
        }
    }
}

// ---------------------------------------------------------------------------
// Windowed attention: one block per (window, head). 49x32 q/k/v, padded to 64.
// S = Q*K^T (WMMA) -> +relpos bias +shift mask -> softmax -> P*V (WMMA).
// ---------------------------------------------------------------------------
__device__ inline int region9(int hs, int ws) {
    int rh = hs < (HH - WS) ? 0 : (hs < (HH - SSH) ? 1 : 2);
    int rw = ws < (HH - WS) ? 0 : (ws < (HH - SSH) ? 1 : 2);
    return rh * 3 + rw;
}

__device__ inline float attn_logit(const float* Ss, const float* rpb,
                                   int n, int m, int head, int wi, int wj) {
    if (m >= NTOK) return -1e30f;
    float s = Ss[n * 64 + m];
    int i1 = n / WS, j1 = n % WS, i2 = m / WS, j2 = m % WS;
    s += rpb[((i1 - i2 + WS - 1) * (2 * WS - 1) + (j1 - j2 + WS - 1)) * NHEAD + head];
    int id1 = region9(wi * WS + i1, wj * WS + j1);
    int id2 = region9(wi * WS + i2, wj * WS + j2);
    if (id1 != id2) s -= 100.0f;
    return s;
}

__global__ __launch_bounds__(256) void attn_kernel(
        const __bf16* __restrict__ qb, const __bf16* __restrict__ kb,
        const __bf16* __restrict__ vb, const float* __restrict__ rpb,
        __bf16* __restrict__ out) {
    int blk = blockIdx.x;
    int win = blk / NHEAD, head = blk % NHEAD;
    int wrem = win & 63, wi = wrem >> 3, wj = wrem & 7;

    __shared__ __bf16 qs[64 * 32];
    __shared__ __bf16 ks[64 * 32];
    __shared__ __bf16 vt[32 * 64];     // V^T: [d][row]
    __shared__ float  Ss[64 * 64];
    __shared__ __bf16 Ps[64 * 64];

    int tid = threadIdx.x;
    size_t base = ((size_t)win * NHEAD + head) * (NTOK * HD);
    __bf16 zb = (__bf16)0.0f;
    for (int idx = tid; idx < 64 * 32; idx += 256) {
        int r = idx >> 5, d = idx & 31;
        bool ok = r < NTOK;
        qs[idx]        = ok ? qb[base + r * HD + d] : zb;
        ks[idx]        = ok ? kb[base + r * HD + d] : zb;
        vt[d * 64 + r] = ok ? vb[base + r * HD + d] : zb;
    }
    __syncthreads();

    int wave = tid >> 5, lane = tid & 31, l15 = lane & 15, hi = lane >> 4;

    // ---- S = Q * K^T : 16 tiles of 16x16, 2 per wave, single K=32 WMMA ----
#pragma unroll
    for (int tt = 0; tt < 2; ++tt) {
        int t = wave * 2 + tt;
        int mt = t >> 2, nt = t & 3;
        v16bf af  = load_frag(qs + (mt * 16 + l15) * 32 + hi * 8);
        v16bf bfr = load_frag(ks + (nt * 16 + l15) * 32 + hi * 8);
        v8f a = {};
        a = wmma_bf16(af, bfr, a);
#pragma unroll
        for (int r = 0; r < 8; ++r)
            Ss[(mt * 16 + hi * 8 + r) * 64 + nt * 16 + l15] = a[r];
    }
    __syncthreads();

    // ---- bias + mask + softmax: 8 rows per wave, 2 cols per lane ----
#pragma unroll
    for (int rr = 0; rr < 8; ++rr) {
        int n = wave * 8 + rr;
        float v0, v1;
        if (n < NTOK) {
            v0 = attn_logit(Ss, rpb, n, lane,      head, wi, wj);
            v1 = attn_logit(Ss, rpb, n, lane + 32, head, wi, wj);
        } else { v0 = v1 = 0.f; }
        float mx = fmaxf(v0, v1);
#pragma unroll
        for (int o = 16; o; o >>= 1) mx = fmaxf(mx, __shfl_xor(mx, o, 32));
        float e0 = __expf(v0 - mx), e1 = __expf(v1 - mx);
        float sm = e0 + e1;
#pragma unroll
        for (int o = 16; o; o >>= 1) sm += __shfl_xor(sm, o, 32);
        float inv = 1.0f / sm;
        Ps[n * 64 + lane]      = (__bf16)(e0 * inv);
        Ps[n * 64 + lane + 32] = (__bf16)(e1 * inv);
    }
    __syncthreads();

    // ---- O = P * V : 8 tiles (4 M x 2 N), 1 per wave, K = 64 (2 steps) ----
    {
        int mt = wave >> 1, nt = wave & 1;
        v8f a = {};
#pragma unroll
        for (int k0 = 0; k0 < 64; k0 += 32) {
            v16bf af  = load_frag(Ps + (mt * 16 + l15) * 64 + k0 + hi * 8);
            v16bf bfr = load_frag(vt + (nt * 16 + l15) * 64 + k0 + hi * 8);
            a = wmma_bf16(af, bfr, a);
        }
#pragma unroll
        for (int r = 0; r < 8; ++r) {
            int m = mt * 16 + hi * 8 + r;
            if (m < NTOK) {
                int d = nt * 16 + l15;
                out[((size_t)win * NTOK + m) * CC + head * HD + d] = (__bf16)a[r];
            }
        }
    }
}

// ---------------------------------------------------------------------------
// Host launcher
// ---------------------------------------------------------------------------
extern "C" void kernel_launch(void* const* d_in, const int* in_sizes, int n_in,
                              void* d_out, int out_size, void* d_ws, size_t ws_size,
                              hipStream_t stream) {
    (void)in_sizes; (void)n_in; (void)out_size; (void)ws_size;
    const float* x      = (const float*)d_in[0];
    const float* n1g    = (const float*)d_in[1];
    const float* n1b    = (const float*)d_in[2];
    const float* qkv_w  = (const float*)d_in[3];
    const float* qkv_b  = (const float*)d_in[4];
    const float* rpb    = (const float*)d_in[5];
    const float* proj_w = (const float*)d_in[6];
    const float* proj_b = (const float*)d_in[7];
    const float* n2g    = (const float*)d_in[8];
    const float* n2b    = (const float*)d_in[9];
    const float* fc1_w  = (const float*)d_in[10];
    const float* fc1_b  = (const float*)d_in[11];
    const float* fc2_w  = (const float*)d_in[12];
    const float* fc2_b  = (const float*)d_in[13];

    char* ws = (char*)d_ws;
    size_t off = 0;
    auto carve = [&](size_t bytes) {
        char* p = ws + off;
        off += (bytes + 255) & ~(size_t)255;
        return p;
    };
    __bf16* hwb     = (__bf16*)carve((size_t)MROWS * CC * 2);
    __bf16* qkvb    = (__bf16*)carve((size_t)3 * QKV_TSZ * 2);
    __bf16* attn_o  = (__bf16*)carve((size_t)MROWS * CC * 2);
    float*  x1      = (float*) carve((size_t)MROWS * CC * 4);
    __bf16* mb      = (__bf16*)carve((size_t)MROWS * CC * 2);
    __bf16* h1      = (__bf16*)carve((size_t)MROWS * 4 * CC * 2);
    __bf16* qkv_wt  = (__bf16*)carve((size_t)CC * 3 * CC * 2);
    __bf16* proj_wt = (__bf16*)carve((size_t)CC * CC * 2);
    __bf16* fc1_wt  = (__bf16*)carve((size_t)CC * 4 * CC * 2);
    __bf16* fc2_wt  = (__bf16*)carve((size_t)4 * CC * CC * 2);

    // 1. weights -> bf16, transposed to N x K
    conv_wt_kernel<<<(CC * 3 * CC + 255) / 256, 256, 0, stream>>>(qkv_w, qkv_wt, CC, 3 * CC);
    conv_wt_kernel<<<(CC * CC + 255) / 256, 256, 0, stream>>>(proj_w, proj_wt, CC, CC);
    conv_wt_kernel<<<(CC * 4 * CC + 255) / 256, 256, 0, stream>>>(fc1_w, fc1_wt, CC, 4 * CC);
    conv_wt_kernel<<<(4 * CC * CC + 255) / 256, 256, 0, stream>>>(fc2_w, fc2_wt, 4 * CC, CC);

    // 2. LN1 + shift + window partition (f32 -> bf16)
    ln_kernel<<<MROWS / 8, 256, 0, stream>>>(x, n1g, n1b, hwb, 1);

    // 3. QKV GEMM (100352x192 @ 192x576), scatter to q/k/v, fold q*scale
    {
        dim3 g(MROWS / 128, (3 * CC) / 64);
        gemm_bf16_kernel<EPI_QKV><<<g, 256, 0, stream>>>(hwb, qkv_wt, qkv_b,
            MROWS, 3 * CC, CC, qkvb, nullptr, nullptr);
    }

    // 4. windowed attention (2048 windows x 6 heads)
    attn_kernel<<<NWINS * NHEAD, 256, 0, stream>>>(qkvb, qkvb + QKV_TSZ,
                                                   qkvb + 2 * (size_t)QKV_TSZ,
                                                   rpb, attn_o);

    // 5. proj GEMM + window reverse + unshift + residual -> x1 (f32)
    {
        dim3 g(MROWS / 128, CC / 64);
        gemm_bf16_kernel<EPI_PROJ><<<g, 256, 0, stream>>>(attn_o, proj_wt, proj_b,
            MROWS, CC, CC, nullptr, x1, x);
    }

    // 6. LN2 (f32 -> bf16)
    ln_kernel<<<MROWS / 8, 256, 0, stream>>>(x1, n2g, n2b, mb, 0);

    // 7. fc1 GEMM + exact GELU -> h1 (bf16)
    {
        dim3 g(MROWS / 128, (4 * CC) / 64);
        gemm_bf16_kernel<EPI_GELU><<<g, 256, 0, stream>>>(mb, fc1_wt, fc1_b,
            MROWS, 4 * CC, CC, h1, nullptr, nullptr);
    }

    // 8. fc2 GEMM + residual -> d_out (f32)
    {
        dim3 g(MROWS / 128, CC / 64);
        gemm_bf16_kernel<EPI_OUT><<<g, 256, 0, stream>>>(h1, fc2_wt, fc2_b,
            MROWS, CC, 4 * CC, nullptr, (float*)d_out, x1);
    }
}
